// Model_sequential_44315472560256
// MI455X (gfx1250) — matrix operand order repaired and checked by reading, not compiled, not run on hardware
//
#include <hip/hip_runtime.h>
#include <hip/hip_bf16.h>
#include <stdint.h>

// ---------------------------------------------------------------------------
// GCN + relative-representation head for MI455X (gfx1250, wave32, WMMA).
// Dense GEMMs use v_wmma_f32_16x16x32_f16 with *pre-converted f16 operands*:
// A matrices are converted f32->f16 once per use (O(M*K)), B operands are
// stored transposed-in-f16 ([N,K], K-major). The GEMM K-loop is then pure:
// 10 global_load_b128 directly into WMMA operand registers + 4 independent
// WMMAs -- no conversions, no branches, no register recycling of loads.
// Scatter-add uses global_atomic_add_f32, one wave per edge, lanes coalesced.
// ---------------------------------------------------------------------------

typedef __attribute__((ext_vector_type(8)))  _Float16 v8h;
typedef __attribute__((ext_vector_type(16))) _Float16 v16h;
typedef __attribute__((ext_vector_type(8)))  float    v8f;

#define D_IN   128
#define D_HID  256
#define N_CLS  16

__device__ __forceinline__ v16h combine(v8h lo, v8h hi) {
    v16h r;
#pragma unroll
    for (int i = 0; i < 8; ++i) { r[i] = lo[i]; r[i + 8] = hi[i]; }
    return r;
}

__device__ __forceinline__ void store_tile(const v8f& acc, float* __restrict__ C,
                                           const float* __restrict__ bias,
                                           int M, int N, int cmb, int cn, int RELU) {
    if (cn >= N) return;                          // after all WMMA: divergence is fine
    const float badd = bias ? bias[cn] : 0.0f;
#pragma unroll
    for (int r = 0; r < 8; ++r) {
        const int cm = cmb + r;
        if (cm < M) {
            float v = acc[r] + badd;
            if (RELU) v = fmaxf(v, 0.0f);
            C[(long)cm * N + cn] = v;
        }
    }
}

// ---------------------------------------------------------------------------
// C[M,N] = act(A[M,K] @ Bt^T + bias)
//   A:  f16 [M,K] row-major.   Bt: f16 [N,K] row-major (B^T, K-major).
// K multiple of 32 (128/256/512 here). One wave computes a 16x64 C strip:
// 4 independent accumulators, A fragment reused 4x per K-step, branch-free
// K loop (invalid sub-tiles compute on clamped rows, stores guarded).
// CDNA5 wave32 fragment layouts (ISA 7.12.2):
//   A (16-bit 16x32): m = lane&15; halves e: K = (e&7) + 16*(e>>3) + 8*(lane>>4)
//   B (16-bit 32x16): n = lane&15; halves e: ISA order as A; code uses K = e + 16*(lane>>4) for A and B
//   C (f32 16x16):    reg r: M = r + 8*(lane>>4), N = lane&15
// ---------------------------------------------------------------------------
__global__ __launch_bounds__(256)
void k_wmma_gemm(const _Float16* __restrict__ A, const _Float16* __restrict__ Bt,
                 const float* __restrict__ bias, float* __restrict__ C,
                 int M, int N, int K, int RELU)
{
    const int lane    = threadIdx.x & 31;
    const int wave    = threadIdx.x >> 5;
    const int tiles_m = (M + 15) >> 4;
    const int tiles_n = (N + 15) >> 4;
    const int ngroups = (tiles_n + 3) >> 2;       // 4 n-tiles per wave
    const int t = blockIdx.x * (blockDim.x >> 5) + wave;
    if (t >= tiles_m * ngroups) return;           // wave-uniform exit
    const int tm  = t / ngroups;
    const int tn0 = (t - tm * ngroups) << 2;

    const int nlane   = lane & 15;
    const int kbase_a = (lane >> 4) << 4;
    const int kbase_b = (lane >> 4) << 4;
    // clamped indices: loads always in-bounds, never predicated
    const int row_a = min(tm * 16 + nlane, M - 1);
    const int col0  = min((tn0 + 0) * 16 + nlane, N - 1);
    const int col1  = min((tn0 + 1) * 16 + nlane, N - 1);
    const int col2  = min((tn0 + 2) * 16 + nlane, N - 1);
    const int col3  = min((tn0 + 3) * 16 + nlane, N - 1);

    const _Float16* __restrict__ arow = A  + (long)row_a * K + kbase_a;
    const _Float16* __restrict__ b0p  = Bt + (long)col0  * K + kbase_b;
    const _Float16* __restrict__ b1p  = Bt + (long)col1  * K + kbase_b;
    const _Float16* __restrict__ b2p  = Bt + (long)col2  * K + kbase_b;
    const _Float16* __restrict__ b3p  = Bt + (long)col3  * K + kbase_b;

    v8f acc0 = {}, acc1 = {}, acc2 = {}, acc3 = {};
    for (int k0 = 0; k0 < K; k0 += 32) {
        // 10x global_load_b128 straight into WMMA operand registers
        const v8h* ap = (const v8h*)(arow + k0);
        const v8h* q0 = (const v8h*)(b0p + k0);
        const v8h* q1 = (const v8h*)(b1p + k0);
        const v8h* q2 = (const v8h*)(b2p + k0);
        const v8h* q3 = (const v8h*)(b3p + k0);
        const v8h alo = ap[0], ahi = ap[1];       // K +0..15 contiguous
        const v8h b0l = q0[0], b0h = q0[1];       // K +0..15 contiguous
        const v8h b1l = q1[0], b1h = q1[1];
        const v8h b2l = q2[0], b2h = q2[1];
        const v8h b3l = q3[0], b3h = q3[1];

        const v16h av  = combine(alo, ahi);
        const v16h bv0 = combine(b0l, b0h);
        const v16h bv1 = combine(b1l, b1h);
        const v16h bv2 = combine(b2l, b2h);
        const v16h bv3 = combine(b3l, b3h);

        acc0 = __builtin_amdgcn_wmma_f32_16x16x32_f16(false, av, false, bv0,
                                                      (short)0, acc0, false, false);
        acc1 = __builtin_amdgcn_wmma_f32_16x16x32_f16(false, av, false, bv1,
                                                      (short)0, acc1, false, false);
        acc2 = __builtin_amdgcn_wmma_f32_16x16x32_f16(false, av, false, bv2,
                                                      (short)0, acc2, false, false);
        acc3 = __builtin_amdgcn_wmma_f32_16x16x32_f16(false, av, false, bv3,
                                                      (short)0, acc3, false, false);
    }

    const int cmb = tm * 16 + ((lane >> 4) << 3);
    const bool u1 = tn0 + 1 < tiles_n, u2 = tn0 + 2 < tiles_n, u3 = tn0 + 3 < tiles_n;
    store_tile(acc0, C, bias, M, N, cmb, (tn0 + 0) * 16 + nlane, RELU);
    if (u1) store_tile(acc1, C, bias, M, N, cmb, (tn0 + 1) * 16 + nlane, RELU);
    if (u2) store_tile(acc2, C, bias, M, N, cmb, (tn0 + 2) * 16 + nlane, RELU);
    if (u3) store_tile(acc3, C, bias, M, N, cmb, (tn0 + 3) * 16 + nlane, RELU);
}

// --- f32 -> f16 elementwise (vectorized x4) ---------------------------------
__global__ void k_f32tof16(const float* __restrict__ S, _Float16* __restrict__ D, long n)
{
    const long i = ((long)blockIdx.x * blockDim.x + threadIdx.x) * 4;
    if (i + 3 < n) {
        const float4 f = *(const float4*)(S + i);
        v8h* dv = nullptr; (void)dv;
        D[i + 0] = (_Float16)f.x; D[i + 1] = (_Float16)f.y;
        D[i + 2] = (_Float16)f.z; D[i + 3] = (_Float16)f.w;
    } else {
        for (long j = i; j < n; ++j) D[j] = (_Float16)S[j];
    }
}

// --- transpose + convert: D[c*R + r] = (f16)S[r*C + c]  (S is f32 [R,C]) ----
__global__ void k_transpose_h(const float* __restrict__ S, _Float16* __restrict__ D,
                              int R, int C)
{
    const long i = (long)blockIdx.x * blockDim.x + threadIdx.x;
    if (i < (long)R * C) {
        const int c = (int)(i / R);
        const int r = (int)(i - (long)c * R);
        D[i] = (_Float16)S[(long)r * C + c];      // coalesced writes
    }
}

// --- degree / normalization -------------------------------------------------
__global__ void k_deg_init(float* deg, int n) {
    int i = blockIdx.x * blockDim.x + threadIdx.x;
    if (i < n) deg[i] = 1.0f;                     // self loop
}
__global__ void k_deg_edges(const int* __restrict__ ei, int E, float* deg) {
    int e = blockIdx.x * blockDim.x + threadIdx.x;
    if (e < E) atomicAdd(&deg[ei[E + e]], 1.0f);  // dst row of edge_index
}
__global__ void k_dis(const float* __restrict__ deg, float* dis, int n) {
    int i = blockIdx.x * blockDim.x + threadIdx.x;
    if (i < n) dis[i] = rsqrtf(deg[i]);
}

// --- H[i,:] = bias (pre-scatter accumulator init), cols = 256 ---------------
__global__ void k_fill_bias(float* __restrict__ H, const float* __restrict__ b, long total) {
    long i = (long)blockIdx.x * blockDim.x + threadIdx.x;
    if (i < total) H[i] = b[i & (D_HID - 1)];
}

// --- scatter: H[dst,:] += norm * T[src,:]; one wave per (edge|self-loop) ----
__global__ __launch_bounds__(256)
void k_scatter(const float* __restrict__ T, const int* __restrict__ ei,
               int E, int n_nodes, const float* __restrict__ dis,
               float* __restrict__ H)
{
    const int w    = (blockIdx.x * blockDim.x + threadIdx.x) >> 5;
    const int lane = threadIdx.x & 31;
    if (w >= E + n_nodes) return;
    int s, d; float norm;
    if (w < E) {
        s = ei[w]; d = ei[E + w];
        norm = dis[s] * dis[d];
    } else {
        s = d = w - E;
        const float v = dis[s];
        norm = v * v;
    }
    const float* __restrict__ src = T + (long)s * D_HID;
    float* __restrict__ dst       = H + (long)d * D_HID;
#pragma unroll
    for (int j = 0; j < D_HID / 32; ++j) {
        const int c = lane + j * 32;              // coalesced across the wave
        atomicAdd(&dst[c], norm * src[c]);
    }
}

__global__ void k_relu(float* H, long n) {
    long i = (long)blockIdx.x * blockDim.x + threadIdx.x;
    if (i < n) H[i] = fmaxf(H[i], 0.0f);
}

// --- row L2 normalize, one wave per 256-wide row ----------------------------
__global__ __launch_bounds__(256)
void k_rownorm(float* __restrict__ H, int rows)
{
    const int w    = (blockIdx.x * blockDim.x + threadIdx.x) >> 5;
    const int lane = threadIdx.x & 31;
    if (w >= rows) return;
    float* r = H + (long)w * D_HID;
    float ss = 0.0f;
#pragma unroll
    for (int j = 0; j < D_HID / 32; ++j) {
        const float v = r[lane + j * 32];
        ss += v * v;
    }
#pragma unroll
    for (int off = 16; off; off >>= 1) ss += __shfl_xor(ss, off, 32);
    const float sc = rsqrtf(ss);
#pragma unroll
    for (int j = 0; j < D_HID / 32; ++j) r[lane + j * 32] *= sc;
}

// --- anchors[p,:] = H[prot[p],:] --------------------------------------------
__global__ void k_gather_anchors(const float* __restrict__ H, const int* __restrict__ prot,
                                 float* __restrict__ A)
{
    A[(long)blockIdx.x * D_HID + threadIdx.x] =
        H[(long)prot[blockIdx.x] * D_HID + threadIdx.x];
}

// --- log_softmax over 16-wide rows; each half-wave owns one row -------------
__global__ __launch_bounds__(256)
void k_logsoftmax16(const float* __restrict__ src, float* __restrict__ dst, int rows)
{
    const int w    = (blockIdx.x * blockDim.x + threadIdx.x) >> 5;
    const int lane = threadIdx.x & 31;
    const int row  = w * 2 + (lane >> 4);
    const int col  = lane & 15;
    if (row >= rows) return;
    const float v = src[(long)row * N_CLS + col];
    float m = v;
#pragma unroll
    for (int off = 8; off; off >>= 1) m = fmaxf(m, __shfl_xor(m, off, 16));
    float s = __expf(v - m);
#pragma unroll
    for (int off = 8; off; off >>= 1) s += __shfl_xor(s, off, 16);
    dst[(long)row * N_CLS + col] = v - m - __logf(s);
}

// ---------------------------------------------------------------------------
static inline int cdiv(long a, long b) { return (int)((a + b - 1) / b); }

extern "C" void kernel_launch(void* const* d_in, const int* in_sizes, int n_in,
                              void* d_out, int out_size, void* d_ws, size_t ws_size,
                              hipStream_t stream)
{
    // inputs (setup_inputs order)
    const float* x    = (const float*)d_in[0];
    const int*   ei   = (const int*)  d_in[1];   // [2,E]
    const int*   prot = (const int*)  d_in[2];   // [P]
    const float* W0   = (const float*)d_in[4];
    const float* b0   = (const float*)d_in[5];
    const float* W1   = (const float*)d_in[6];
    const float* b1   = (const float*)d_in[7];
    const float* Wl1  = (const float*)d_in[8];
    const float* bl1  = (const float*)d_in[9];
    const float* Wl2  = (const float*)d_in[10];
    const float* bl2  = (const float*)d_in[11];

    const int Nn = in_sizes[0] / D_IN;           // 50000
    const int E  = in_sizes[1] / 2;              // 800000
    const int P  = in_sizes[2];                  // 512

    // workspace layout (large, aligned buffers first)
    char* ws = (char*)d_ws;
    float*    T      = (float*)ws;     ws += (size_t)Nn * D_HID * 4;   // messages
    float*    H      = (float*)ws;     ws += (size_t)Nn * D_HID * 4;   // accum / hidden
    _Float16* Ah     = (_Float16*)ws;  ws += (size_t)Nn * 512 * 2;     // f16 A staging (max Nn x 512)
    float*    deg    = (float*)ws;     ws += (size_t)Nn * 4;
    float*    dis    = (float*)ws;     ws += (size_t)Nn * 4;
    float*    anch   = (float*)ws;     ws += (size_t)P * D_HID * 4;
    float*    z1     = (float*)ws;     ws += (size_t)P * D_HID * 4;
    float*    z2     = (float*)ws;     ws += (size_t)P * N_CLS * 4;
    _Float16* anchH  = (_Float16*)ws;  ws += (size_t)P * D_HID * 2;    // anchors f16 (A and Bt roles)
    _Float16* z1H    = (_Float16*)ws;  ws += (size_t)P * D_HID * 2;
    _Float16* W0h    = (_Float16*)ws;  ws += (size_t)D_IN  * D_HID * 2;  // [256,128] f16
    _Float16* W1h    = (_Float16*)ws;  ws += (size_t)D_HID * D_HID * 2;  // [256,256] f16
    _Float16* Wl1h   = (_Float16*)ws;  ws += (size_t)D_HID * D_HID * 2;  // [256,256] f16
    _Float16* Wl2h   = (_Float16*)ws;  ws += (size_t)D_HID * N_CLS * 2;  // [16,256]  f16
    _Float16* oprTh  = (_Float16*)ws;  ws += (size_t)P * N_CLS * 2;      // [16,512]  f16

    // outputs: out [Nn,16] ++ x_rel [Nn,P] ++ out_proto [P,16]
    float* out_f  = (float*)d_out;
    float* xrel   = out_f + (size_t)Nn * N_CLS;
    float* oproto = xrel  + (size_t)Nn * P;

    const int TB = 256, WPB = 8;                 // 8 wave32s per block
    const long NH = (long)Nn * D_HID;

    auto gemm_grid = [](int M, int N) {
        const int tiles = ((M + 15) / 16) * (((N + 15) / 16 + 3) / 4);
        return cdiv(tiles, 8);
    };

    // 0) weights -> transposed f16 ([N,K], K-major)
    k_transpose_h<<<cdiv((long)D_IN  * D_HID, TB), TB, 0, stream>>>(W0,  W0h,  D_IN,  D_HID);
    k_transpose_h<<<cdiv((long)D_HID * D_HID, TB), TB, 0, stream>>>(W1,  W1h,  D_HID, D_HID);
    k_transpose_h<<<cdiv((long)D_HID * D_HID, TB), TB, 0, stream>>>(Wl1, Wl1h, D_HID, D_HID);
    k_transpose_h<<<cdiv((long)D_HID * N_CLS, TB), TB, 0, stream>>>(Wl2, Wl2h, D_HID, N_CLS);

    // 1) degree + rsqrt
    k_deg_init <<<cdiv(Nn, TB), TB, 0, stream>>>(deg, Nn);
    k_deg_edges<<<cdiv(E,  TB), TB, 0, stream>>>(ei, E, deg);
    k_dis      <<<cdiv(Nn, TB), TB, 0, stream>>>(deg, dis, Nn);

    // 2) layer 0: T = x@W0 ; H = b0 + scatter(norm*T[src]->dst) ; relu
    k_f32tof16 <<<cdiv((long)Nn * D_IN / 4, TB), TB, 0, stream>>>(x, Ah, (long)Nn * D_IN);
    k_wmma_gemm<<<gemm_grid(Nn, D_HID), TB, 0, stream>>>(Ah, W0h, nullptr, T,
                                                         Nn, D_HID, D_IN, 0);
    k_fill_bias<<<cdiv(NH, TB), TB, 0, stream>>>(H, b0, NH);
    k_scatter  <<<cdiv((long)(E + Nn), WPB), TB, 0, stream>>>(T, ei, E, Nn, dis, H);
    k_relu     <<<cdiv(NH, TB), TB, 0, stream>>>(H, NH);

    // 3) layer 1: T = H@W1 ; H = b1 + scatter
    k_f32tof16 <<<cdiv(NH / 4, TB), TB, 0, stream>>>(H, Ah, NH);
    k_wmma_gemm<<<gemm_grid(Nn, D_HID), TB, 0, stream>>>(Ah, W1h, nullptr, T,
                                                         Nn, D_HID, D_HID, 0);
    k_fill_bias<<<cdiv(NH, TB), TB, 0, stream>>>(H, b1, NH);
    k_scatter  <<<cdiv((long)(E + Nn), WPB), TB, 0, stream>>>(T, ei, E, Nn, dis, H);

    // 4) row L2-normalize; gather anchors (rows unit norm -> cosine = dot)
    k_rownorm       <<<cdiv(Nn, WPB), TB, 0, stream>>>(H, Nn);
    k_gather_anchors<<<P, D_HID, 0, stream>>>(H, prot, anch);
    k_f32tof16      <<<cdiv((long)P * D_HID / 4, TB), TB, 0, stream>>>(anch, anchH, (long)P * D_HID);

    // 5) prototype head: out_proto = log_softmax(relu(anchors@Wl1+bl1)@Wl2+bl2)
    k_wmma_gemm<<<gemm_grid(P, D_HID), TB, 0, stream>>>(anchH, Wl1h, bl1, z1,
                                                        P, D_HID, D_HID, 1);
    k_f32tof16 <<<cdiv((long)P * D_HID / 4, TB), TB, 0, stream>>>(z1, z1H, (long)P * D_HID);
    k_wmma_gemm<<<gemm_grid(P, N_CLS), TB, 0, stream>>>(z1H, Wl2h, bl2, z2,
                                                        P, N_CLS, D_HID, 0);
    k_logsoftmax16<<<cdiv((P + 1) / 2, WPB), TB, 0, stream>>>(z2, oproto, P);
    k_transpose_h <<<cdiv((long)P * N_CLS, TB), TB, 0, stream>>>(oproto, oprTh, P, N_CLS);

    // 6) x_rel = H @ anchors^T (anchH is [N,K] f16; unit rows -> denom = 1)
    k_f32tof16 <<<cdiv(NH / 4, TB), TB, 0, stream>>>(H, Ah, NH);   // post-norm H
    k_wmma_gemm<<<gemm_grid(Nn, P), TB, 0, stream>>>(Ah, anchH, nullptr, xrel,
                                                     Nn, P, D_HID, 0);

    // 7) out = log_softmax(x_rel @ out_proto)
    k_f32tof16 <<<cdiv((long)Nn * P / 4, TB), TB, 0, stream>>>(xrel, Ah, (long)Nn * P);
    k_wmma_gemm<<<gemm_grid(Nn, N_CLS), TB, 0, stream>>>(Ah, oprTh, nullptr, out_f,
                                                         Nn, N_CLS, P, 0);
    k_logsoftmax16<<<cdiv((Nn + 1) / 2, WPB), TB, 0, stream>>>(out_f, out_f, Nn);

    (void)n_in; (void)out_size; (void)ws_size;
}
